// _PointnetSAModuleBase_30468497998037
// MI455X (gfx1250) — matrix operand order repaired and checked by reading, not compiled, not run on hardware
//
#include <hip/hip_runtime.h>
#include <hip/hip_bf16.h>

// Problem constants (from reference)
#define B_   8
#define N_   8192
#define S_   1024
#define NS_  32
#define C_   64
#define R2_  0.04f        // RADIUS^2
#define EPS_ 1e-5f
#define CNT_ 262144.0f    // B*S*NS

typedef __attribute__((ext_vector_type(16))) __bf16 v16bf;
typedef __attribute__((ext_vector_type(8)))  float  v8f;

union BF16Frag { v16bf v; unsigned short u[16]; };

__device__ __forceinline__ unsigned short f2bf(float f) {
  unsigned u = __float_as_uint(f);
  u += 0x7fffu + ((u >> 16) & 1u);     // round-to-nearest-even
  return (unsigned short)(u >> 16);
}

__device__ __forceinline__ v8f wmma_bf16(v16bf a, v16bf b, v8f c) {
  // D(16x16 f32) = A(16x32 bf16) * B(32x16 bf16) + C
  return __builtin_amdgcn_wmma_f32_16x16x32_bf16(false, a, false, b, (short)0, c,
                                                 false, false);
}

// A fragment (weights, bf16 row-major OxK). A 16x32 in this code's contiguous k order (matches B loaders):
// lanes 0-15 (M=lane): K 0..15; lanes 16-31: K 16..31.
__device__ __forceinline__ v16bf load_a(const unsigned short* __restrict__ W,
                                        int Kdim, int rt, int kk, int lane) {
  BF16Frag r;
  const int m  = rt * 16 + (lane & 15);
  const int hi = lane >> 4;
  const unsigned short* wr = W + m * Kdim + kk * 32;
#pragma unroll
  for (int e = 0; e < 16; e++) {
    int koff = e + (hi ? 16 : 0);
    r.u[e] = wr[koff];
  }
  return r.v;
}

// B fragment from stored y (K x 32, k-major stride 32) with BN+ReLU applied.
// B 32x16 in this code's contiguous k order (matches load_a): lanes 0-15 hold K 0..15 (N=lane), lanes 16-31 hold K 16..31.
__device__ __forceinline__ v16bf load_b_bn(const float* __restrict__ ybs,
                                           const float* __restrict__ sc,
                                           const float* __restrict__ sh,
                                           int ct, int kk, int lane) {
  BF16Frag r;
  const int n  = ct * 16 + (lane & 15);
  const int kb = kk * 32 + ((lane >> 4) * 16);
#pragma unroll
  for (int e = 0; e < 16; e++) {
    int k = kb + e;
    float v = ybs[k * 32 + n];
    v = fmaxf(0.f, v * sc[k] + sh[k]);
    r.u[e] = f2bf(v);
  }
  return r.v;
}

// Layer-1 B fragment: rows 0..2 = gathered xyz - center, 3..66 = gathered feature,
// 67..95 = zero pad.
__device__ __forceinline__ v16bf load_b_x1(const float* __restrict__ xyzb,
                                           const float* __restrict__ feab,
                                           const int* __restrict__ idxp,
                                           const float* __restrict__ ctr,
                                           int ct, int kk, int lane) {
  BF16Frag r;
  const int n  = ct * 16 + (lane & 15);
  const int pt = idxp[n];
  const int kb = kk * 32 + ((lane >> 4) * 16);
#pragma unroll
  for (int e = 0; e < 16; e++) {
    int k = kb + e;
    float v;
    if (k < 3)       v = xyzb[pt * 3 + k] - ctr[k];
    else if (k < 67) v = feab[(size_t)(k - 3) * N_ + pt];
    else             v = 0.f;
    r.u[e] = f2bf(v);
  }
  return r.v;
}

// ---------------------------------------------------------------- FPS ------
// One workgroup per batch. Each thread keeps 8 points in registers.
__global__ __launch_bounds__(1024) void k_fps(const float* __restrict__ xyz,
                                              int* __restrict__ sidx) {
  const int b   = blockIdx.x;
  const int tid = threadIdx.x;
  const float* bx = xyz + (size_t)b * N_ * 3;
  float px[8], py[8], pz[8], mind[8];
#pragma unroll
  for (int k = 0; k < 8; k++) {
    int p = tid + k * 1024;
    px[k] = bx[p * 3 + 0]; py[k] = bx[p * 3 + 1]; pz[k] = bx[p * 3 + 2];
    mind[k] = 1e10f;
  }
  __shared__ float curx, cury, curz;
  __shared__ float rv[32];
  __shared__ int   ri[32];
  __shared__ int   nxt;
  if (tid == 0) { sidx[b * S_] = 0; curx = px[0]; cury = py[0]; curz = pz[0]; }
  __syncthreads();

  for (int t = 1; t < S_; t++) {
    const float cx = curx, cy = cury, cz = curz;
    float bv = -1.0f; int bi = 0;
#pragma unroll
    for (int k = 0; k < 8; k++) {
      float dx = px[k] - cx, dy = py[k] - cy, dz = pz[k] - cz;
      float d = dx * dx + dy * dy + dz * dz;
      mind[k] = fminf(mind[k], d);
      if (mind[k] > bv) { bv = mind[k]; bi = tid + k * 1024; }
    }
#pragma unroll
    for (int off = 16; off > 0; off >>= 1) {
      float ov = __shfl_xor(bv, off);
      int   oi = __shfl_xor(bi, off);
      if (ov > bv || (ov == bv && oi < bi)) { bv = ov; bi = oi; }
    }
    const int wave = tid >> 5, lane = tid & 31;
    if (lane == 0) { rv[wave] = bv; ri[wave] = bi; }
    __syncthreads();
    if (wave == 0) {
      bv = rv[lane]; bi = ri[lane];
#pragma unroll
      for (int off = 16; off > 0; off >>= 1) {
        float ov = __shfl_xor(bv, off);
        int   oi = __shfl_xor(bi, off);
        if (ov > bv || (ov == bv && oi < bi)) { bv = ov; bi = oi; }
      }
      if (lane == 0) { nxt = bi; sidx[b * S_ + t] = bi; }
    }
    __syncthreads();
    const int nx = nxt;
    if (tid == (nx & 1023)) {
      int k = nx >> 10; curx = px[k]; cury = py[k]; curz = pz[k];
    }
    __syncthreads();
  }
}

// -------------------------------------------------- gather sampled points --
__global__ void k_gather(const float* __restrict__ xyz, const int* __restrict__ sidx,
                         float* __restrict__ nxyz, float* __restrict__ out_xyz,
                         float* __restrict__ out_sidx) {
  int i = blockIdx.x * 256 + threadIdx.x;              // over B*S
  if (i >= B_ * S_) return;
  int b = i >> 10;
  int p = sidx[i];
#pragma unroll
  for (int c = 0; c < 3; c++) {
    float v = xyz[((size_t)b * N_ + p) * 3 + c];
    nxyz[i * 3 + c] = v;
    out_xyz[i * 3 + c] = v;
  }
  out_sidx[i] = (float)p;
}

// ------------------------------------------------------------ ball query ---
// One wave32 per (b,s): ordered ballot scan reproduces sort-based semantics.
__global__ __launch_bounds__(256) void k_ballquery(const float* __restrict__ xyz,
                                                   const float* __restrict__ nxyz,
                                                   int* __restrict__ bidx) {
  const int wid  = (blockIdx.x * 256 + (int)threadIdx.x) >> 5;
  const int lane = threadIdx.x & 31;
  const int b = wid >> 10;
  const float* bx = xyz + (size_t)b * N_ * 3;
  const float cx = nxyz[wid * 3 + 0], cy = nxyz[wid * 3 + 1], cz = nxyz[wid * 3 + 2];
  int* ob = bidx + (size_t)wid * NS_;

  int count = 0, first = 0;
  for (int base = 0; base < N_; base += 32) {
    int p = base + lane;
    float dx = bx[p * 3 + 0] - cx, dy = bx[p * 3 + 1] - cy, dz = bx[p * 3 + 2] - cz;
    bool pred = (dx * dx + dy * dy + dz * dz) < R2_;
    unsigned mask = (unsigned)__ballot(pred);
    if (mask) {
      if (count == 0) first = base + (__ffs(mask) - 1);
      int pos = count + __popc(mask & ((1u << lane) - 1u));
      if (pred && pos < NS_) ob[pos] = p;
      count += __popc(mask);
      if (count >= NS_) break;
    }
  }
  int cnt = count < NS_ ? count : NS_;
  if (lane >= cnt) ob[lane] = (count == 0) ? 0 : first;
}

// ------------------------------------------------------------ conv layers --
// One wave32 per (b,s) group: y(Ox32) = W(OxK) @ x(Kx32) via bf16 WMMA.

__global__ __launch_bounds__(256) void k_conv1(
    const float* __restrict__ xyz, const float* __restrict__ feat,
    const float* __restrict__ nxyz, const int* __restrict__ bidx,
    const unsigned short* __restrict__ W1p, float* __restrict__ y1,
    float* __restrict__ stats) {
  const int wid  = blockIdx.x * 8 + ((int)threadIdx.x >> 5);
  const int lane = threadIdx.x & 31;
  const int b    = wid >> 10;
  const float* xyzb = xyz + (size_t)b * N_ * 3;
  const float* feab = feat + (size_t)b * C_ * N_;
  const int*   idxp = bidx + (size_t)wid * NS_;
  const float* ctr  = nxyz + (size_t)wid * 3;

  const v8f vz = {0.f, 0.f, 0.f, 0.f, 0.f, 0.f, 0.f, 0.f};
  v8f c[4][2];
#pragma unroll
  for (int rt = 0; rt < 4; rt++) { c[rt][0] = vz; c[rt][1] = vz; }

#pragma unroll
  for (int kk = 0; kk < 3; kk++) {                 // K = 96 (67 padded)
    v16bf bf0 = load_b_x1(xyzb, feab, idxp, ctr, 0, kk, lane);
    v16bf bf1 = load_b_x1(xyzb, feab, idxp, ctr, 1, kk, lane);
#pragma unroll
    for (int rt = 0; rt < 4; rt++) {
      v16bf a = load_a(W1p, 96, rt, kk, lane);
      c[rt][0] = wmma_bf16(a, bf0, c[rt][0]);
      c[rt][1] = wmma_bf16(a, bf1, c[rt][1]);
    }
  }

  float* yb = y1 + (size_t)wid * 64 * 32;
  const int nl = lane & 15;
  const int mh = (lane < 16) ? 0 : 8;
#pragma unroll
  for (int rt = 0; rt < 4; rt++) {
#pragma unroll
    for (int r = 0; r < 8; r++) {
      int m = rt * 16 + mh + r;
      yb[m * 32 + nl]      = c[rt][0][r];
      yb[m * 32 + 16 + nl] = c[rt][1][r];
      float sv = c[rt][0][r] + c[rt][1][r];
      float sq = c[rt][0][r] * c[rt][0][r] + c[rt][1][r] * c[rt][1][r];
#pragma unroll
      for (int off = 1; off < 16; off <<= 1) {
        sv += __shfl_xor(sv, off, 16);
        sq += __shfl_xor(sq, off, 16);
      }
      if (nl == 0) { atomicAdd(&stats[m], sv); atomicAdd(&stats[64 + m], sq); }
    }
  }
}

__global__ __launch_bounds__(256) void k_conv2(
    const float* __restrict__ y1, const float* __restrict__ sc1,
    const float* __restrict__ sh1, const unsigned short* __restrict__ W2p,
    float* __restrict__ y2, float* __restrict__ stats) {
  const int wid  = blockIdx.x * 8 + ((int)threadIdx.x >> 5);
  const int lane = threadIdx.x & 31;
  const float* ybs = y1 + (size_t)wid * 64 * 32;

  const v8f vz = {0.f, 0.f, 0.f, 0.f, 0.f, 0.f, 0.f, 0.f};
  v8f c[4][2];
#pragma unroll
  for (int rt = 0; rt < 4; rt++) { c[rt][0] = vz; c[rt][1] = vz; }

#pragma unroll
  for (int kk = 0; kk < 2; kk++) {                 // K = 64
    v16bf bf0 = load_b_bn(ybs, sc1, sh1, 0, kk, lane);
    v16bf bf1 = load_b_bn(ybs, sc1, sh1, 1, kk, lane);
#pragma unroll
    for (int rt = 0; rt < 4; rt++) {
      v16bf a = load_a(W2p, 64, rt, kk, lane);
      c[rt][0] = wmma_bf16(a, bf0, c[rt][0]);
      c[rt][1] = wmma_bf16(a, bf1, c[rt][1]);
    }
  }

  float* yb = y2 + (size_t)wid * 64 * 32;
  const int nl = lane & 15;
  const int mh = (lane < 16) ? 0 : 8;
#pragma unroll
  for (int rt = 0; rt < 4; rt++) {
#pragma unroll
    for (int r = 0; r < 8; r++) {
      int m = rt * 16 + mh + r;
      yb[m * 32 + nl]      = c[rt][0][r];
      yb[m * 32 + 16 + nl] = c[rt][1][r];
      float sv = c[rt][0][r] + c[rt][1][r];
      float sq = c[rt][0][r] * c[rt][0][r] + c[rt][1][r] * c[rt][1][r];
#pragma unroll
      for (int off = 1; off < 16; off <<= 1) {
        sv += __shfl_xor(sv, off, 16);
        sq += __shfl_xor(sq, off, 16);
      }
      if (nl == 0) { atomicAdd(&stats[128 + m], sv); atomicAdd(&stats[192 + m], sq); }
    }
  }
}

// Layer-3 stats pass: compute y3 in registers only, accumulate sum/sumsq.
__global__ __launch_bounds__(256) void k_conv3s(
    const float* __restrict__ y2, const float* __restrict__ sc2,
    const float* __restrict__ sh2, const unsigned short* __restrict__ W3p,
    float* __restrict__ stats) {
  const int wid  = blockIdx.x * 8 + ((int)threadIdx.x >> 5);
  const int lane = threadIdx.x & 31;
  const float* ybs = y2 + (size_t)wid * 64 * 32;

  const v8f vz = {0.f, 0.f, 0.f, 0.f, 0.f, 0.f, 0.f, 0.f};
  v8f c[8][2];
#pragma unroll
  for (int rt = 0; rt < 8; rt++) { c[rt][0] = vz; c[rt][1] = vz; }

#pragma unroll
  for (int kk = 0; kk < 2; kk++) {
    v16bf bf0 = load_b_bn(ybs, sc2, sh2, 0, kk, lane);
    v16bf bf1 = load_b_bn(ybs, sc2, sh2, 1, kk, lane);
#pragma unroll
    for (int rt = 0; rt < 8; rt++) {               // O = 128
      v16bf a = load_a(W3p, 64, rt, kk, lane);
      c[rt][0] = wmma_bf16(a, bf0, c[rt][0]);
      c[rt][1] = wmma_bf16(a, bf1, c[rt][1]);
    }
  }

  const int nl = lane & 15;
  const int mh = (lane < 16) ? 0 : 8;
#pragma unroll
  for (int rt = 0; rt < 8; rt++) {
#pragma unroll
    for (int r = 0; r < 8; r++) {
      int m = rt * 16 + mh + r;
      float sv = c[rt][0][r] + c[rt][1][r];
      float sq = c[rt][0][r] * c[rt][0][r] + c[rt][1][r] * c[rt][1][r];
#pragma unroll
      for (int off = 1; off < 16; off <<= 1) {
        sv += __shfl_xor(sv, off, 16);
        sq += __shfl_xor(sq, off, 16);
      }
      if (nl == 0) { atomicAdd(&stats[256 + m], sv); atomicAdd(&stats[384 + m], sq); }
    }
  }
}

// Final: recompute y3 (cheap), BN3 + ReLU, max over the 32 neighbors.
__global__ __launch_bounds__(256) void k_final(
    const float* __restrict__ y2, const float* __restrict__ sc2,
    const float* __restrict__ sh2, const unsigned short* __restrict__ W3p,
    const float* __restrict__ sc3, const float* __restrict__ sh3,
    float* __restrict__ outf) {
  const int wid  = blockIdx.x * 8 + ((int)threadIdx.x >> 5);
  const int lane = threadIdx.x & 31;
  const int b = wid >> 10;
  const int s = wid & (S_ - 1);
  const float* ybs = y2 + (size_t)wid * 64 * 32;

  const v8f vz = {0.f, 0.f, 0.f, 0.f, 0.f, 0.f, 0.f, 0.f};
  v8f c[8][2];
#pragma unroll
  for (int rt = 0; rt < 8; rt++) { c[rt][0] = vz; c[rt][1] = vz; }

#pragma unroll
  for (int kk = 0; kk < 2; kk++) {
    v16bf bf0 = load_b_bn(ybs, sc2, sh2, 0, kk, lane);
    v16bf bf1 = load_b_bn(ybs, sc2, sh2, 1, kk, lane);
#pragma unroll
    for (int rt = 0; rt < 8; rt++) {
      v16bf a = load_a(W3p, 64, rt, kk, lane);
      c[rt][0] = wmma_bf16(a, bf0, c[rt][0]);
      c[rt][1] = wmma_bf16(a, bf1, c[rt][1]);
    }
  }

  const int nl = lane & 15;
  const int mh = (lane < 16) ? 0 : 8;
#pragma unroll
  for (int rt = 0; rt < 8; rt++) {
#pragma unroll
    for (int r = 0; r < 8; r++) {
      int m = rt * 16 + mh + r;
      float s3 = sc3[m], h3 = sh3[m];
      float v = fmaxf(fmaxf(0.f, c[rt][0][r] * s3 + h3),
                      fmaxf(0.f, c[rt][1][r] * s3 + h3));
#pragma unroll
      for (int off = 1; off < 16; off <<= 1) v = fmaxf(v, __shfl_xor(v, off, 16));
      if (nl == 0) outf[((size_t)b * 128 + m) * S_ + s] = v;
    }
  }
}

// ----------------------------------------------------------- small utils ---
__global__ void k_zero(float* __restrict__ p) {
  int i = blockIdx.x * 256 + threadIdx.x;
  if (i < 512) p[i] = 0.f;
}

__global__ void k_prep(const float* __restrict__ W1, const float* __restrict__ W2,
                       const float* __restrict__ W3, unsigned short* __restrict__ W1p,
                       unsigned short* __restrict__ W2p, unsigned short* __restrict__ W3p) {
  int i = blockIdx.x * 256 + threadIdx.x;
  if (i < 64 * 96) { int o = i / 96, k = i % 96; W1p[i] = f2bf(k < 67 ? W1[o * 67 + k] : 0.f); }
  if (i < 64 * 64) W2p[i] = f2bf(W2[i]);
  if (i < 128 * 64) W3p[i] = f2bf(W3[i]);
}

__global__ void k_finalize(const float* __restrict__ sum, const float* __restrict__ sq,
                           const float* __restrict__ g, const float* __restrict__ bb,
                           float* __restrict__ sc, float* __restrict__ sh, int O) {
  int o = blockIdx.x * blockDim.x + threadIdx.x;
  if (o >= O) return;
  float m = sum[o] * (1.f / CNT_);
  float v = sq[o] * (1.f / CNT_) - m * m;
  float s = g[o] * rsqrtf(v + EPS_);
  sc[o] = s;
  sh[o] = bb[o] - m * s;
}

// ------------------------------------------------------------------ launch -
extern "C" void kernel_launch(void* const* d_in, const int* in_sizes, int n_in,
                              void* d_out, int out_size, void* d_ws, size_t ws_size,
                              hipStream_t stream) {
  (void)in_sizes; (void)n_in; (void)out_size; (void)ws_size;
  const float* xyz  = (const float*)d_in[0];
  const float* feat = (const float*)d_in[1];
  const float* W1 = (const float*)d_in[2];
  const float* g1 = (const float*)d_in[3];
  const float* b1 = (const float*)d_in[4];
  const float* W2 = (const float*)d_in[5];
  const float* g2 = (const float*)d_in[6];
  const float* b2 = (const float*)d_in[7];
  const float* W3 = (const float*)d_in[8];
  const float* g3 = (const float*)d_in[9];
  const float* b3 = (const float*)d_in[10];

  char* ws = (char*)d_ws;
  size_t off = 0;
  auto take = [&](size_t bytes) -> char* {
    char* p = ws + off;
    off = (off + bytes + 255) & ~(size_t)255;
    return p;
  };
  int*            sidx  = (int*)take((size_t)B_ * S_ * 4);
  float*          nxyz  = (float*)take((size_t)B_ * S_ * 3 * 4);
  int*            bidx  = (int*)take((size_t)B_ * S_ * NS_ * 4);
  unsigned short* W1p   = (unsigned short*)take(64 * 96 * 2);
  unsigned short* W2p   = (unsigned short*)take(64 * 64 * 2);
  unsigned short* W3p   = (unsigned short*)take(128 * 64 * 2);
  float*          stats = (float*)take(512 * 4);   // sum1,sq1,sum2,sq2,sum3(128),sq3(128)
  float*          bn    = (float*)take(512 * 4);   // sc1,sh1,sc2,sh2,sc3(128),sh3(128)
  float*          y1    = (float*)take((size_t)B_ * S_ * 64 * NS_ * 4);  // 64 MB
  float*          y2    = (float*)take((size_t)B_ * S_ * 64 * NS_ * 4);  // 64 MB

  float* out_xyz  = (float*)d_out;
  float* out_feat = out_xyz + (size_t)B_ * S_ * 3;
  float* out_sidx = out_feat + (size_t)B_ * 128 * S_;

  k_prep<<<32, 256, 0, stream>>>(W1, W2, W3, W1p, W2p, W3p);
  k_zero<<<2, 256, 0, stream>>>(stats);
  k_fps<<<B_, 1024, 0, stream>>>(xyz, sidx);
  k_gather<<<32, 256, 0, stream>>>(xyz, sidx, nxyz, out_xyz, out_sidx);
  k_ballquery<<<1024, 256, 0, stream>>>(xyz, nxyz, bidx);

  k_conv1<<<1024, 256, 0, stream>>>(xyz, feat, nxyz, bidx, W1p, y1, stats);
  k_finalize<<<1, 64, 0, stream>>>(stats + 0, stats + 64, g1, b1, bn + 0, bn + 64, 64);

  k_conv2<<<1024, 256, 0, stream>>>(y1, bn + 0, bn + 64, W2p, y2, stats);
  k_finalize<<<1, 64, 0, stream>>>(stats + 128, stats + 192, g2, b2, bn + 128, bn + 192, 64);

  k_conv3s<<<1024, 256, 0, stream>>>(y2, bn + 128, bn + 192, W3p, stats);
  k_finalize<<<1, 128, 0, stream>>>(stats + 256, stats + 384, g3, b3, bn + 256, bn + 384, 128);

  k_final<<<1024, 256, 0, stream>>>(y2, bn + 128, bn + 192, W3p, bn + 256, bn + 384, out_feat);
}
